// M_90829968376485
// MI455X (gfx1250) — matrix operand order repaired and checked by reading, not compiled, not run on hardware
//
#include <hip/hip_runtime.h>
#include <math.h>

// Problem constants (from reference): query/key (B, C, T) fp32 -> out (B, K, T) fp32
#define B_    8
#define C_    1024
#define T_    4096
#define KWIN  127
#define PAD_  63

// Tiling
#define NT       32            // t columns per block
#define MSPAN    160           // key-position rows per block (158 used, padded to 10 tiles of 16)
#define MT       10            // m tiles (16 rows each)
#define CCH      32            // channel chunk staged in LDS per iteration (one K=32 WMMA)
#define NCHUNK   (C_ / CCH)    // 32 chunks
#define KP       40            // LDS row pitch in bf16 elems (80B rows: 16B-aligned)
#define GP       33            // LDS gram pitch in floats (pad to dodge bank conflicts)
#define NTHREADS 640           // 20 wave32 waves: one 16x16 output tile per wave (10 x 2 tiles)

typedef __attribute__((ext_vector_type(16))) __bf16 v16bf;
typedef __attribute__((ext_vector_type(8)))  __bf16 v8bf;
typedef __attribute__((ext_vector_type(8)))  float  v8f;

// Pack two fp32 into two bf16 in one dword.
// Prefer the HW packed convert (v_cvt_pk_bf16_f32) if the toolchain declares it;
// otherwise round-to-nearest (ties away) + a single v_perm_b32 splice (3 VALU ops).
__device__ __forceinline__ unsigned pack_bf16(float a, float b) {
#if __has_builtin(__builtin_amdgcn_cvt_pk_bf16_f32)
    typedef __attribute__((ext_vector_type(2))) __bf16 v2bf;
    v2bf r = __builtin_amdgcn_cvt_pk_bf16_f32(a, b);
    unsigned pk;
    __builtin_memcpy(&pk, &r, 4);
    return pk;
#else
    union { float f; unsigned u; } x, y;
    x.f = a; y.f = b;
    unsigned ar = x.u + 0x8000u;
    unsigned br = y.u + 0x8000u;
    // bytes 0-1 from S1(=ar) bytes {2,3}; bytes 2-3 from S0(=br) bytes {2,3}
    return __builtin_amdgcn_perm(br, ar, 0x07060302u);
#endif
}

__global__ __launch_bounds__(NTHREADS)
void band_attn_softmax_kernel(const float* __restrict__ q,
                              const float* __restrict__ k,
                              float* __restrict__ out) {
    __shared__ __bf16 sK[2][MSPAN * KP];   // key window, [m][c] bf16, double-buffered
    __shared__ __bf16 sQ[2][NT * KP];      // query tile, [t][c] bf16, double-buffered
    __shared__ float  sG[MSPAN * GP];      // gram band, [m][t] f32
    __shared__ float  sMax[20][NT];
    __shared__ float  sSum[20][NT];

    const int tid  = threadIdx.x;
    const int lane = tid & 31;
    const int wave = tid >> 5;
    const int b    = blockIdx.x / (T_ / NT);
    const int t0   = (blockIdx.x % (T_ / NT)) * NT;

    const int mt   = wave % MT;         // which 16-row key tile
    const int nt   = wave / MT;         // which 16-col t tile
    const int half = lane >> 4;
    const int l16  = lane & 15;
    const int arow = mt * 16 + l16;     // key-span row this lane reads for A
    const int bcol = nt * 16 + l16;     // t column this lane reads for B

    const float* qb = q + (size_t)b * C_ * T_;
    const float* kb = k + (size_t)b * C_ * T_;

    // ---- chunk-invariant staging coordinates ----
    // key: 640 threads = 4 groups x 160 rows; 4 units cover 16 channel-pairs
    const int  m0      = tid % MSPAN;          // key row (consecutive lanes -> coalesced)
    const int  cq      = tid / MSPAN;          // 0..3
    const int  kp0     = t0 - PAD_ + m0;       // absolute key position (chunk invariant)
    const bool kvalid  = (kp0 >= 0) && (kp0 < T_);
    // query: threads 0..511 = 32 t x 16 channel-pairs
    const bool qactive = tid < NT * (CCH / 2);
    const int  tq      = tid & (NT - 1);
    const int  cpq     = tid >> 5;             // 0..15 when active

    float kv0[4], kv1[4], qv0 = 0.f, qv1 = 0.f;

    auto preload = [&](int c0) {               // global -> registers (next chunk)
        #pragma unroll
        for (int i = 0; i < 4; ++i) {
            kv0[i] = 0.f; kv1[i] = 0.f;
            if (kvalid) {
                const float* src = kb + (size_t)(c0 + 2 * (i * 4 + cq)) * T_ + kp0;
                kv0[i] = src[0];
                kv1[i] = src[T_];
            }
        }
        if (qactive) {
            const float* src = qb + (size_t)(c0 + 2 * cpq) * T_ + (t0 + tq);
            qv0 = __builtin_nontemporal_load(src);        // query: single-use, keep L2 for key
            qv1 = __builtin_nontemporal_load(src + T_);
        }
    };
    auto stage = [&](int buf) {                // registers -> LDS (bf16 pack)
        #pragma unroll
        for (int i = 0; i < 4; ++i) {
            *(unsigned*)(&sK[buf][m0 * KP + 2 * (i * 4 + cq)]) = pack_bf16(kv0[i], kv1[i]);
        }
        if (qactive) {
            *(unsigned*)(&sQ[buf][tq * KP + 2 * cpq]) = pack_bf16(qv0, qv1);
        }
    };

    v8f acc = {};
    auto mma = [&](int buf) {
        v16bf a, bm;
        // A (16x32 bf16): lane<16 holds K 0..15; lane>=16 holds K 16..31
        *((v8bf*)&a)       = *(const v8bf*)(&sK[buf][arow * KP + half * 16]);
        *(((v8bf*)&a) + 1) = *(const v8bf*)(&sK[buf][arow * KP + half * 16 + 8]);
        // B (32x16 bf16): lane<16 holds K 0..15; lane>=16 holds K 16..31
        *((v8bf*)&bm)       = *(const v8bf*)(&sQ[buf][bcol * KP + half * 16]);
        *(((v8bf*)&bm) + 1) = *(const v8bf*)(&sQ[buf][bcol * KP + half * 16 + 8]);
        acc = __builtin_amdgcn_wmma_f32_16x16x32_bf16(false, a, false, bm,
                                                      (short)0, acc, false, false);
    };

    // ---- software-pipelined main loop: 1 barrier per chunk, loads overlap WMMA ----
    preload(0);
    stage(0);
    __syncthreads();
    for (int c = 0; c < NCHUNK; ++c) {
        const int buf = c & 1;
        if (c + 1 < NCHUNK) preload((c + 1) * CCH);   // issue global loads first
        mma(buf);                                     // consume current buffer
        if (c + 1 < NCHUNK) stage(buf ^ 1);           // fill the other buffer
        __syncthreads();
    }

    // ---- spill gram tile: acc[i] = gram[mt*16 + half*8 + i][nt*16 + l16] ----
    #pragma unroll
    for (int i = 0; i < 8; ++i) {
        sG[(mt * 16 + half * 8 + i) * GP + (nt * 16 + l16)] = acc[i];
    }
    __syncthreads();

    // ---- masked softmax over the 127-wide window; part = wave (20 parts of <=7 k's), col = lane ----
    const int part = wave;
    const int col  = lane;
    float mx = -INFINITY;
    #pragma unroll
    for (int j = 0; j < 7; ++j) {
        int kidx = part * 7 + j;
        if (kidx < KWIN) {
            int p = t0 + col + kidx - PAD_;
            if (p >= 0 && p < T_) mx = fmaxf(mx, sG[(col + kidx) * GP + col]);
        }
    }
    sMax[part][col] = mx;
    __syncthreads();
    float mtot = -INFINITY;
    #pragma unroll
    for (int j = 0; j < 20; ++j) mtot = fmaxf(mtot, sMax[j][col]);

    float s = 0.f;
    #pragma unroll
    for (int j = 0; j < 7; ++j) {
        int kidx = part * 7 + j;
        if (kidx < KWIN) {
            int p = t0 + col + kidx - PAD_;
            if (p >= 0 && p < T_) {
                float e = __expf(sG[(col + kidx) * GP + col] - mtot);
                sG[(col + kidx) * GP + col] = e;     // cache exp for the store pass (own entry)
                s += e;
            }
        }
    }
    sSum[part][col] = s;
    __syncthreads();
    float stot = 0.f;
    #pragma unroll
    for (int j = 0; j < 20; ++j) stot += sSum[j][col];
    const float inv = 1.0f / stot;

    float* ob = out + (size_t)b * KWIN * T_ + (t0 + col);
    #pragma unroll
    for (int j = 0; j < 7; ++j) {
        int kidx = part * 7 + j;
        if (kidx < KWIN) {
            int p = t0 + col + kidx - PAD_;
            float v = (p >= 0 && p < T_) ? sG[(col + kidx) * GP + col] * inv : 0.0f;
            __builtin_nontemporal_store(v, &ob[(size_t)kidx * T_]);  // write-once, coalesced
        }
    }
}

extern "C" void kernel_launch(void* const* d_in, const int* in_sizes, int n_in,
                              void* d_out, int out_size, void* d_ws, size_t ws_size,
                              hipStream_t stream) {
    const float* q = (const float*)d_in[0];
    const float* k = (const float*)d_in[1];
    float* out = (float*)d_out;
    (void)in_sizes; (void)n_in; (void)out_size; (void)d_ws; (void)ws_size;

    dim3 grid(B_ * (T_ / NT));   // 8 * 128 = 1024 blocks: one (batch, t-tile) each
    dim3 block(NTHREADS);        // 20 wave32 waves
    hipLaunchKernelGGL(band_attn_softmax_kernel, grid, block, 0, stream, q, k, out);
}
